// MultiHeadAttn_12309376271014
// MI455X (gfx1250) — matrix operand order repaired and checked by reading, not compiled, not run on hardware
//
#include <hip/hip_runtime.h>
#include <hip/hip_bf16.h>

typedef _Float16 f16;
typedef __attribute__((ext_vector_type(16))) _Float16 v16h;
typedef __attribute__((ext_vector_type(8)))  float    v8f;
typedef __attribute__((ext_vector_type(4)))  unsigned int u32x4;
typedef __attribute__((ext_vector_type(8)))  int      i32x8;
typedef __attribute__((ext_vector_type(4)))  int      i32x4;

#define B_   4
#define S_   2048
#define H_   16
#define DM_  1024
#define BS_  (B_ * S_)

struct F16Frag { uint4 lo, hi; };
static_assert(sizeof(v16h) == 32, "v16h must be 32 bytes");
static_assert(sizeof(F16Frag) == 32, "frag pack must be 32 bytes");

__device__ __forceinline__ v16h pack_frag(uint4 lo, uint4 hi) {
  F16Frag t{lo, hi};
  return __builtin_bit_cast(v16h, t);
}

// A-matrix 16x32 f16 fragment from row-major [16 x ld] tile.
// ISA layout: lanes 0-15 row=lane, elems 0..7 = K0..7, elems 8..15 = K16..23;
//             lanes 16-31 row=lane-16, K8..15 then K24..31.
__device__ __forceinline__ v16h load_a_frag(const f16* p, int ld) {
  const int l   = threadIdx.x & 31;
  const int row = l & 15;
  const int k0  = (l >> 4) << 3;               // 0 or 8
  const f16* r  = p + (size_t)row * ld + k0;
  uint4 lo = *(const uint4*)(r);
  uint4 hi = *(const uint4*)(r + 16);
  return pack_frag(lo, hi);
}

// B-matrix 32x16 f16 fragment, source stored transposed: rows = N, cols = K, stride ld.
// ISA layout: lanes 0-15 col=lane, K0..7 then K16..23; lanes 16-31 col=lane-16, K8..15 then K24..31.
__device__ __forceinline__ v16h load_bT_frag(const f16* p, int ld) {
  const int l  = threadIdx.x & 31;
  const int n  = l & 15;
  const int k0 = (l >> 4) << 3;                // 0 or 8
  const f16* r = p + (size_t)n * ld + k0;
  uint4 lo = *(const uint4*)(r);
  uint4 hi = *(const uint4*)(r + 16);
  return pack_frag(lo, hi);
}

__device__ __forceinline__ v8f wmma_f16(v16h a, v16h b, v8f c) {
  return __builtin_amdgcn_wmma_f32_16x16x32_f16(false, a, false, b, (short)0, c,
                                                false, false);
}

// TDM: DMA one 16-row x 32-col f16 tile (row stride = stride_elems) into LDS.
// D# layout per CDNA5 ISA ch.8: group0 {count,lds,global,type=2},
// group1 {data_size=2B, dims, tile 32x16, dim0 stride}; groups 2/3 zero (2-D).
// This toolchain exposes the 6-arg builtin: (g0 v4u32, g1 v8i32, v4i32, v4i32, v8i32, cpol).
__device__ __forceinline__ void tdm_load_tile_16x32(unsigned int lds_off,
                                                    unsigned long long gaddr,
                                                    unsigned int stride_elems) {
  u32x4 g0;
  g0.x = 1u;                                            // count=1, user descriptor
  g0.y = lds_off;                                       // lds_addr (bytes)
  g0.z = (unsigned int)(gaddr & 0xffffffffu);           // global_addr[31:0]
  g0.w = (unsigned int)((gaddr >> 32) & 0x1ffffffu)     // global_addr[56:32]
         | (2u << 30);                                  // type = 2 ("image")
  i32x8 g1;
  g1[0] = 0x00010000;                                   // data_size=1 (2 bytes)
  g1[1] = (int)(stride_elems << 16);                    // tensor_dim0[15:0]
  g1[2] = (int)((stride_elems >> 16) | (16u << 16));    // dim0 hi | tensor_dim1=16
  g1[3] = (int)(32u << 16);                             // tile_dim0 = 32
  g1[4] = 16;                                           // tile_dim1 = 16, tile_dim2 = 0
  g1[5] = (int)stride_elems;                            // tensor_dim0_stride lo32
  g1[6] = 0;                                            // stride hi | dim1_stride lo
  g1[7] = 0;
  i32x4 z4 = {0, 0, 0, 0};
  i32x8 z8 = {0, 0, 0, 0, 0, 0, 0, 0};
  __builtin_amdgcn_tensor_load_to_lds(g0, g1, z4, z4, z8, 0);
}

// reductions within each 16-lane half (rows of a C fragment live on one half)
__device__ __forceinline__ float hmax16(float x) {
  #pragma unroll
  for (int m = 1; m <= 8; m <<= 1) x = fmaxf(x, __shfl_xor(x, m, 32));
  return x;
}
__device__ __forceinline__ float hsum16(float x) {
  #pragma unroll
  for (int m = 1; m <= 8; m <<= 1) x += __shfl_xor(x, m, 32);
  return x;
}

// ---------------- prep kernels ----------------
__global__ void cvt_f16_kernel(const float* __restrict__ in, f16* __restrict__ out, int n) {
  int i = blockIdx.x * 256 + threadIdx.x;
  if (i < n) out[i] = (f16)in[i];
}

// out[j*1024 + i] = (f16) in[i*1024 + j]   (1024x1024)
__global__ void transpose_cvt_kernel(const float* __restrict__ in, f16* __restrict__ out) {
  int o = blockIdx.x * 256 + threadIdx.x;
  int j = o >> 10;
  int i = o & 1023;
  out[o] = (f16)in[(size_t)i * DM_ + j];
}

// ---------------- QKV projection GEMM ----------------
// Each wave: 16 rows of X x one head's 64 output cols. Wt is [out_col][k] f16.
// A-tile streamed by TDM into double-buffered LDS; B frags from L2-resident weights.
// vmode==0: store [B,H,S,64]; vmode==1: store V transposed [B,H,64,S].
__global__ __launch_bounds__(256)
void qkv_gemm_kernel(const f16* __restrict__ Xh, const f16* __restrict__ Wt,
                     const float* __restrict__ bias, f16* __restrict__ out, int vmode) {
  __shared__ __align__(128) f16 abuf[8][2][16 * 32];   // 16 KB: per-wave double buffer
  const int wloc = threadIdx.x >> 5;
  const int wave = (blockIdx.x * 256 + threadIdx.x) >> 5;
  const int l    = threadIdx.x & 31;
  const int g    = wave & (H_ - 1);            // head
  const int mt   = wave >> 4;                  // 16-row tile of X
  const int r0   = mt * 16;

  const unsigned int lds0 = (unsigned int)(uintptr_t)&abuf[wloc][0][0];
  const unsigned int lds1 = (unsigned int)(uintptr_t)&abuf[wloc][1][0];
  const unsigned long long ga = (unsigned long long)(uintptr_t)(Xh + (size_t)r0 * DM_);

  tdm_load_tile_16x32(lds0, ga, DM_);          // prime buffer 0 (kk = 0)

  v8f acc[4] = {{}, {}, {}, {}};
  int buf = 0;
  for (int kk = 0; kk < DM_; kk += 32) {
    if (kk + 32 < DM_) {
      // previous reads of the other buffer must retire before TDM refills it
      asm volatile("s_wait_dscnt 0" ::: "memory");
      tdm_load_tile_16x32(buf ? lds0 : lds1, ga + (size_t)(kk + 32) * 2, DM_);
      __builtin_amdgcn_s_wait_tensorcnt((short)1);   // current tile ready
    } else {
      __builtin_amdgcn_s_wait_tensorcnt((short)0);
    }
    v16h a = load_a_frag(&abuf[wloc][buf][0], 32);
    #pragma unroll
    for (int t = 0; t < 4; ++t) {
      v16h b = load_bT_frag(Wt + (size_t)(g * 64 + t * 16) * DM_ + kk, DM_);
      acc[t] = wmma_f16(a, b, acc[t]);
    }
    buf ^= 1;
  }

  const int bidx = r0 / S_;
  const int s0   = r0 % S_;
  const int mrow = (l >> 4) << 3;
  #pragma unroll
  for (int t = 0; t < 4; ++t) {
    const int dcol = t * 16 + (l & 15);
    const float bj = bias[g * 64 + dcol];
    #pragma unroll
    for (int v = 0; v < 8; ++v) {
      const float val = acc[t][v] + bj;
      const int s = s0 + mrow + v;
      size_t idx;
      if (vmode) idx = ((size_t)(bidx * H_ + g) * 64 + dcol) * S_ + s;
      else       idx = ((size_t)(bidx * H_ + g) * S_ + s) * 64 + dcol;
      out[idx] = (f16)val;
    }
  }
}

// ---------------- flash attention ----------------
// Q,K: [B,H,S,64] f16.  Vt: [B,H,64,S] f16.  A out: [B,S,H*64] f16.
__global__ __launch_bounds__(256)
void attn_kernel(const f16* __restrict__ Q, const f16* __restrict__ K,
                 const f16* __restrict__ Vt, f16* __restrict__ A) {
  __shared__ __align__(16) f16 pl[8][16][32];  // per-wave P tile for C->A relayout

  const int w    = threadIdx.x >> 5;
  const int l    = threadIdx.x & 31;
  const int wave = blockIdx.x * 8 + w;
  const int qt   = wave & (S_ / 16 - 1);       // 128 q-tiles per (b,h)
  const int bh   = wave >> 7;
  const int h    = bh & (H_ - 1);
  const int b    = bh >> 4;
  const int q0   = qt * 16;
  const int n    = l & 15;
  const int mrow = (l >> 4) << 3;

  const f16* qptr = Q + ((size_t)(b * H_ + h) * S_ + q0) * 64;
  const v16h aq0 = load_a_frag(qptr, 64);       // dims 0..31
  const v16h aq1 = load_a_frag(qptr + 32, 64);  // dims 32..63

  v8f o0 = {}, o1 = {}, o2 = {}, o3 = {};
  float mi[8], li[8];
  #pragma unroll
  for (int v = 0; v < 8; ++v) { mi[v] = -__builtin_inff(); li[v] = 0.0f; }

  const f16* kbaseptr = K  + (size_t)(b * H_ + h) * S_ * 64;
  const f16* vbase    = Vt + (size_t)(b * H_ + h) * 64 * S_;
  const int n_kt = (q0 + 15) / 32 + 1;         // causal: skip tiles past diagonal

  for (int kt = 0; kt < n_kt; ++kt) {
    const int kb = kt * 32;
    const f16* kp = kbaseptr + (size_t)kb * 64;

    // scores: S = Q Kt, 16x32 tile as two 16x16 C fragments
    v8f s0v = {}, s1v = {};
    v16h b0 = load_bT_frag(kp, 64);
    v16h b1 = load_bT_frag(kp + 16 * 64, 64);
    s0v = wmma_f16(aq0, b0, s0v);
    s1v = wmma_f16(aq0, b1, s1v);
    v16h b2 = load_bT_frag(kp + 32, 64);
    v16h b3 = load_bT_frag(kp + 16 * 64 + 32, 64);
    s0v = wmma_f16(aq1, b2, s0v);
    s1v = wmma_f16(aq1, b3, s1v);

    // scale, causal mask, online softmax
    const int k0abs = kb + n;
    const int k1abs = kb + 16 + n;
    float p0[8], p1[8];
    #pragma unroll
    for (int v = 0; v < 8; ++v) {
      const int qabs = q0 + mrow + v;
      float x0 = s0v[v] * 0.125f;
      float x1 = s1v[v] * 0.125f;
      if (k0abs > qabs) x0 = -__builtin_inff();
      if (k1abs > qabs) x1 = -__builtin_inff();
      const float tmax = hmax16(fmaxf(x0, x1));
      const float nm   = fmaxf(mi[v], tmax);
      const float corr = __expf(mi[v] - nm);
      const float e0 = __expf(x0 - nm);
      const float e1 = __expf(x1 - nm);
      const float ts = hsum16(e0 + e1);
      li[v] = li[v] * corr + ts;
      mi[v] = nm;
      o0[v] *= corr; o1[v] *= corr; o2[v] *= corr; o3[v] *= corr;
      p0[v] = e0; p1[v] = e1;
    }

    // relayout P (C-frag) -> A-frag through LDS
    #pragma unroll
    for (int v = 0; v < 8; ++v) {
      pl[w][mrow + v][n]      = (f16)p0[v];
      pl[w][mrow + v][16 + n] = (f16)p1[v];
    }
    asm volatile("s_wait_dscnt 0" ::: "memory");
    v16h ap;
    {
      const int row = l & 15;
      const int k0  = (l >> 4) << 3;
      uint4 lo = *(const uint4*)&pl[w][row][k0];
      uint4 hi = *(const uint4*)&pl[w][row][k0 + 16];
      ap = pack_frag(lo, hi);
    }

    // O += P @ V  (B frags contiguous from transposed V)
    const f16* vp = vbase + kb;
    v16h bv0 = load_bT_frag(vp, S_);
    v16h bv1 = load_bT_frag(vp + (size_t)16 * S_, S_);
    v16h bv2 = load_bT_frag(vp + (size_t)32 * S_, S_);
    v16h bv3 = load_bT_frag(vp + (size_t)48 * S_, S_);
    o0 = wmma_f16(ap, bv0, o0);
    o1 = wmma_f16(ap, bv1, o1);
    o2 = wmma_f16(ap, bv2, o2);
    o3 = wmma_f16(ap, bv3, o3);
  }

  // normalize and store attn output [B,S,H*64] f16
  f16* ao = A + (size_t)(b * S_ + q0) * DM_ + h * 64;
  #pragma unroll
  for (int v = 0; v < 8; ++v) {
    const float inv = 1.0f / li[v];
    const size_t rbase = (size_t)(mrow + v) * DM_;
    ao[rbase + n]      = (f16)(o0[v] * inv);
    ao[rbase + 16 + n] = (f16)(o1[v] * inv);
    ao[rbase + 32 + n] = (f16)(o2[v] * inv);
    ao[rbase + 48 + n] = (f16)(o3[v] * inv);
  }
}

// ---------------- output projection ----------------
__global__ __launch_bounds__(256)
void oproj_kernel(const f16* __restrict__ A, const f16* __restrict__ Wot,
                  const float* __restrict__ bo, float* __restrict__ out) {
  __shared__ __align__(128) f16 abuf[8][2][16 * 32];   // 16 KB: per-wave double buffer
  const int wloc = threadIdx.x >> 5;
  const int wave = (blockIdx.x * 256 + threadIdx.x) >> 5;
  const int l    = threadIdx.x & 31;
  const int g    = wave & 15;                  // 64-col group
  const int mt   = wave >> 4;
  const int r0   = mt * 16;

  const unsigned int lds0 = (unsigned int)(uintptr_t)&abuf[wloc][0][0];
  const unsigned int lds1 = (unsigned int)(uintptr_t)&abuf[wloc][1][0];
  const unsigned long long ga = (unsigned long long)(uintptr_t)(A + (size_t)r0 * DM_);

  tdm_load_tile_16x32(lds0, ga, DM_);

  v8f acc[4] = {{}, {}, {}, {}};
  int buf = 0;
  for (int kk = 0; kk < DM_; kk += 32) {
    if (kk + 32 < DM_) {
      asm volatile("s_wait_dscnt 0" ::: "memory");
      tdm_load_tile_16x32(buf ? lds0 : lds1, ga + (size_t)(kk + 32) * 2, DM_);
      __builtin_amdgcn_s_wait_tensorcnt((short)1);
    } else {
      __builtin_amdgcn_s_wait_tensorcnt((short)0);
    }
    v16h a = load_a_frag(&abuf[wloc][buf][0], 32);
    #pragma unroll
    for (int t = 0; t < 4; ++t) {
      v16h b = load_bT_frag(Wot + (size_t)(g * 64 + t * 16) * DM_ + kk, DM_);
      acc[t] = wmma_f16(a, b, acc[t]);
    }
    buf ^= 1;
  }

  const int mrow = (l >> 4) << 3;
  #pragma unroll
  for (int t = 0; t < 4; ++t) {
    const int col  = g * 64 + t * 16 + (l & 15);
    const float bj = bo[col];
    #pragma unroll
    for (int v = 0; v < 8; ++v)
      out[(size_t)(r0 + mrow + v) * DM_ + col] = acc[t][v] + bj;
  }
}

extern "C" void kernel_launch(void* const* d_in, const int* in_sizes, int n_in,
                              void* d_out, int out_size, void* d_ws, size_t ws_size,
                              hipStream_t stream) {
  (void)in_sizes; (void)n_in; (void)out_size; (void)ws_size;
  const float* X  = (const float*)d_in[0];
  // d_in[1] = attn_mask: causal mask is computed analytically, not loaded
  const float* Wq = (const float*)d_in[2];
  const float* bq = (const float*)d_in[3];
  const float* Wk = (const float*)d_in[4];
  const float* bk = (const float*)d_in[5];
  const float* Wv = (const float*)d_in[6];
  const float* bv = (const float*)d_in[7];
  const float* Wo = (const float*)d_in[8];
  const float* bo = (const float*)d_in[9];
  float* out = (float*)d_out;

  char* ws = (char*)d_ws;
  const size_t MB = 1ull << 20;
  f16* Xh  = (f16*)(ws + 0 * MB);    // 16 MB  X in f16
  f16* Wqt = (f16*)(ws + 16 * MB);   //  2 MB  Wq^T f16
  f16* Wkt = (f16*)(ws + 18 * MB);
  f16* Wvt = (f16*)(ws + 20 * MB);
  f16* Wot = (f16*)(ws + 22 * MB);
  f16* Qd  = (f16*)(ws + 24 * MB);   // 16 MB  [B,H,S,64]
  f16* Kd  = (f16*)(ws + 40 * MB);   // 16 MB  [B,H,S,64]
  f16* Vtd = (f16*)(ws + 56 * MB);   // 16 MB  [B,H,64,S]
  f16* Ad  = (f16*)(ws + 72 * MB);   // 16 MB  attn out [B,S,1024]  (total 88 MB)

  cvt_f16_kernel<<<(BS_ * DM_) / 256, 256, 0, stream>>>(X, Xh, BS_ * DM_);
  transpose_cvt_kernel<<<(DM_ * DM_) / 256, 256, 0, stream>>>(Wq, Wqt);
  transpose_cvt_kernel<<<(DM_ * DM_) / 256, 256, 0, stream>>>(Wk, Wkt);
  transpose_cvt_kernel<<<(DM_ * DM_) / 256, 256, 0, stream>>>(Wv, Wvt);
  transpose_cvt_kernel<<<(DM_ * DM_) / 256, 256, 0, stream>>>(Wo, Wot);

  const int gemm_blocks = (BS_ / 16) * H_ / 8;        // 1024 blocks x 8 waves
  qkv_gemm_kernel<<<gemm_blocks, 256, 0, stream>>>(Xh, Wqt, bq, Qd, 0);
  qkv_gemm_kernel<<<gemm_blocks, 256, 0, stream>>>(Xh, Wkt, bk, Kd, 0);
  qkv_gemm_kernel<<<gemm_blocks, 256, 0, stream>>>(Xh, Wvt, bv, Vtd, 1);

  const int attn_blocks = (B_ * H_ * (S_ / 16)) / 8;  // 1024 blocks
  attn_kernel<<<attn_blocks, 256, 0, stream>>>(Qd, Kd, Vtd, Ad);

  oproj_kernel<<<gemm_blocks, 256, 0, stream>>>(Ad, Wot, bo, out);
}
